// GraphormerAttnLayer_62989990363553
// MI455X (gfx1250) — matrix operand order repaired and checked by reading, not compiled, not run on hardware
//
#include <hip/hip_runtime.h>
#include <math.h>

// Problem constants (match reference)
#define NN   16384
#define EE   768
#define HH   12
#define DHD  64
#define BBG  64
#define J3E  2304   // 3*E
#define HIDN 16

typedef __attribute__((ext_vector_type(16))) _Float16 v16h;
typedef __attribute__((ext_vector_type(8)))  _Float16 v8h;
typedef __attribute__((ext_vector_type(8)))  float    v8f;

static __device__ inline int imin(int a, int b) { return a < b ? a : b; }

static __device__ inline v8f zero8f() {
  v8f z;
#pragma unroll
  for (int i = 0; i < 8; ++i) z[i] = 0.0f;
  return z;
}

// A-fragment (16x32 f16, MxK): lane L holds row (L&15); halves 0..15 = K[k0..k0+15],
// k0 = (L<16)?0:16.  p points at row + kc + k0.
static __device__ inline v16h load_afrag(const _Float16* p) {
  v8h lo = *(const v8h*)p;
  v8h hi = *(const v8h*)(p + 8);
  v16h a;
#pragma unroll
  for (int i = 0; i < 8; ++i) { a[i] = lo[i]; a[i + 8] = hi[i]; }
  return a;
}

// B-fragment (32x16 f16, KxN): lane L holds col (L&15); 16 contiguous K values
// starting at kb = (L<16)?0:16.  p points at B-col's contiguous K run.
static __device__ inline v16h load_bfrag(const _Float16* p) {
  return *(const v16h*)p;
}

static __device__ inline v8f wmma_f16(v16h a, v16h b, v8f c) {
  return __builtin_amdgcn_wmma_f32_16x16x32_f16(false, a, false, b, (short)0, c,
                                                false, false);
}

// ---------------------------------------------------------------------------
// 1) starts/counts via binary search over sorted `batch`
// ---------------------------------------------------------------------------
__global__ __launch_bounds__(64) void k_starts(const int* __restrict__ batch,
                                               int n, int* __restrict__ starts,
                                               int* __restrict__ counts) {
  __shared__ int sh[BBG + 1];
  int b = threadIdx.x;
  if (b < BBG) {
    int lo = 0, hi = n;
    while (lo < hi) {
      int mid = (lo + hi) >> 1;
      if (batch[mid] < b) lo = mid + 1; else hi = mid;
    }
    sh[b] = lo;
  }
  if (b == 0) sh[BBG] = n;
  __syncthreads();
  if (b < BBG) { starts[b] = sh[b]; counts[b] = sh[b + 1] - sh[b]; }
}

// ---------------------------------------------------------------------------
// 2) LayerNorm -> f16 activations
// ---------------------------------------------------------------------------
__global__ __launch_bounds__(256) void k_ln(const float* __restrict__ x,
                                            const float* __restrict__ w,
                                            const float* __restrict__ bln,
                                            _Float16* __restrict__ xn) {
  int row = blockIdx.x, tid = threadIdx.x;
  const float* xr = x + (size_t)row * EE;
  float v0 = xr[tid], v1 = xr[tid + 256], v2 = xr[tid + 512];
  float s = v0 + v1 + v2, ss = v0 * v0 + v1 * v1 + v2 * v2;
  __shared__ float rs[256], rss[256];
  rs[tid] = s; rss[tid] = ss;
  __syncthreads();
  for (int o = 128; o > 0; o >>= 1) {
    if (tid < o) { rs[tid] += rs[tid + o]; rss[tid] += rss[tid + o]; }
    __syncthreads();
  }
  float mean = rs[0] * (1.0f / EE);
  float var  = rss[0] * (1.0f / EE) - mean * mean;
  float inv  = rsqrtf(var + 1e-5f);
  _Float16* xo = xn + (size_t)row * EE;
  xo[tid]       = (_Float16)(((v0 - mean) * inv) * w[tid]       + bln[tid]);
  xo[tid + 256] = (_Float16)(((v1 - mean) * inv) * w[tid + 256] + bln[tid + 256]);
  xo[tid + 512] = (_Float16)(((v2 - mean) * inv) * w[tid + 512] + bln[tid + 512]);
}

// ---------------------------------------------------------------------------
// 3) f32 -> f16 conversion (weights)
// ---------------------------------------------------------------------------
__global__ __launch_bounds__(256) void k_f2h(const float* __restrict__ a,
                                             _Float16* __restrict__ o, int n) {
  for (int i = blockIdx.x * blockDim.x + threadIdx.x; i < n;
       i += gridDim.x * blockDim.x)
    o[i] = (_Float16)a[i];
}

// ---------------------------------------------------------------------------
// 4) C[n,j] = sum_k A[n,k] * W[j,k] + bias[j]   (A f16 NxK, W f16 JxK, C f16)
//    Block = 4 waves; each wave: 64 rows x 64 cols (4x4 accum fragments) so
//    every B (weight) fragment load feeds 4 WMMAs.
//    grid.x = N/64, grid.y = J/256
// ---------------------------------------------------------------------------
__global__ __launch_bounds__(128) void k_gemm_bias(
    const _Float16* __restrict__ A, const _Float16* __restrict__ W,
    const float* __restrict__ bias, _Float16* __restrict__ C, int K, int J) {
  int lane = threadIdx.x & 31;
  int wv   = threadIdx.x >> 5;           // 0..3
  int row0 = blockIdx.x * 64;
  int jb   = (blockIdx.y * 4 + wv) * 64;
  int nl   = lane & 15;
  bool hih = lane >= 16;
  int k0 = hih ? 16 : 0, kb = hih ? 16 : 0;

  v8f acc[4][4];
#pragma unroll
  for (int mt = 0; mt < 4; ++mt)
#pragma unroll
    for (int nt = 0; nt < 4; ++nt) acc[mt][nt] = zero8f();

  for (int kc = 0; kc < K; kc += 32) {
    v16h a[4];
#pragma unroll
    for (int mt = 0; mt < 4; ++mt)
      a[mt] = load_afrag(A + (size_t)(row0 + mt * 16 + nl) * K + kc + k0);
#pragma unroll
    for (int nt = 0; nt < 4; ++nt) {
      int j = jb + nt * 16 + nl;
      v16h b = load_bfrag(W + (size_t)j * K + kc + kb);
#pragma unroll
      for (int mt = 0; mt < 4; ++mt)
        acc[mt][nt] = wmma_f16(a[mt], b, acc[mt][nt]);
    }
  }
#pragma unroll
  for (int nt = 0; nt < 4; ++nt) {
    int j = jb + nt * 16 + nl;
    float bj = bias[j];
#pragma unroll
    for (int mt = 0; mt < 4; ++mt)
#pragma unroll
      for (int r = 0; r < 8; ++r) {
        int m = row0 + mt * 16 + r + (hih ? 8 : 0);
        C[(size_t)m * J + j] = (_Float16)(acc[mt][nt][r] + bj);
      }
  }
}

// ---------------------------------------------------------------------------
// 5) Flash attention: 1 wave per (qtile16, head, graph). qkv layout [N,2304]:
//    q = cols 0..767, k = 768..1535, v = 1536..2303 (per-head stride 64).
// ---------------------------------------------------------------------------
__global__ __launch_bounds__(32) void k_attn(const _Float16* __restrict__ qkv,
                                             const int* __restrict__ starts,
                                             const int* __restrict__ counts,
                                             _Float16* __restrict__ oatt) {
  int qt = blockIdx.x, h = blockIdx.y, g = blockIdx.z;
  int L = counts[g];
  if (qt * 16 >= L) return;
  int s = starts[g];

  int lane = threadIdx.x;
  int nl = lane & 15;
  bool hih = lane >= 16;
  int k0 = hih ? 16 : 0, kb = hih ? 16 : 0;

  // Q fragments (DH=64 -> 2 k-chunks of 32)
  int q = qt * 16 + nl;
  int nq = s + imin(q, L - 1);
  v16h aq[2];
#pragma unroll
  for (int kc = 0; kc < 2; ++kc)
    aq[kc] = load_afrag(qkv + (size_t)nq * J3E + h * DHD + kc * 32 + k0);

  float mrow[8], lrow[8];
  v8f o[4];
#pragma unroll
  for (int r = 0; r < 8; ++r) { mrow[r] = -1e30f; lrow[r] = 0.0f; }
#pragma unroll
  for (int j = 0; j < 4; ++j) o[j] = zero8f();

  __shared__ __align__(32) _Float16 Ps[16][32];

  int nkt = (L + 31) / 32;
  for (int kt = 0; kt < nkt; ++kt) {
    // S = Q * K^T  (two 16x16 tiles covering 32 keys)
    v8f s0 = zero8f(), s1 = zero8f();
#pragma unroll
    for (int nt = 0; nt < 2; ++nt) {
      int key = kt * 32 + nt * 16 + nl;
      int nk  = s + imin(key, L - 1);
      const _Float16* kp = qkv + (size_t)nk * J3E + EE + h * DHD + kb;
#pragma unroll
      for (int kc = 0; kc < 2; ++kc) {
        v16h b = load_bfrag(kp + kc * 32);
        if (nt == 0) s0 = wmma_f16(aq[kc], b, s0);
        else         s1 = wmma_f16(aq[kc], b, s1);
      }
    }
    // scale + key mask (mask depends only on this lane's key column)
    bool m0 = (kt * 32 + nl) >= L;
    bool m1 = (kt * 32 + 16 + nl) >= L;
#pragma unroll
    for (int r = 0; r < 8; ++r) {
      s0[r] = m0 ? -1e9f : s0[r] * 0.125f;
      s1[r] = m1 ? -1e9f : s1[r] * 0.125f;
    }
    // row max across the 16 lanes of this half
    float tm[8];
#pragma unroll
    for (int r = 0; r < 8; ++r) tm[r] = fmaxf(s0[r], s1[r]);
#pragma unroll
    for (int m = 1; m < 16; m <<= 1)
#pragma unroll
      for (int r = 0; r < 8; ++r) tm[r] = fmaxf(tm[r], __shfl_xor(tm[r], m, 32));
    // online softmax rescale
#pragma unroll
    for (int r = 0; r < 8; ++r) {
      float mn = fmaxf(mrow[r], tm[r]);
      float sc = __expf(mrow[r] - mn);
      mrow[r] = mn;
      lrow[r] *= sc;
#pragma unroll
      for (int j = 0; j < 4; ++j) o[j][r] *= sc;
    }
    // p = exp(s - m), row-sum
    float rsum[8];
#pragma unroll
    for (int r = 0; r < 8; ++r) {
      s0[r] = __expf(s0[r] - mrow[r]);
      s1[r] = __expf(s1[r] - mrow[r]);
      rsum[r] = s0[r] + s1[r];
    }
#pragma unroll
    for (int m = 1; m < 16; m <<= 1)
#pragma unroll
      for (int r = 0; r < 8; ++r) rsum[r] += __shfl_xor(rsum[r], m, 32);
#pragma unroll
    for (int r = 0; r < 8; ++r) lrow[r] += rsum[r];

    // C-layout -> A-layout via LDS
#pragma unroll
    for (int r = 0; r < 8; ++r) {
      int m = r + (hih ? 8 : 0);
      Ps[m][nl]      = (_Float16)s0[r];
      Ps[m][16 + nl] = (_Float16)s1[r];
    }
    __syncthreads();
    v16h ap = load_afrag(&Ps[nl][k0]);

    // O += P * V
    size_t rowbase[16];
#pragma unroll
    for (int t = 0; t < 16; ++t) {
      int key = kt * 32 + kb + t;
      int nk  = s + imin(key, L - 1);
      rowbase[t] = (size_t)nk * J3E + 2 * EE + h * DHD + nl;
    }
#pragma unroll
    for (int j = 0; j < 4; ++j) {
      v16h bv;
#pragma unroll
      for (int t = 0; t < 16; ++t) bv[t] = qkv[rowbase[t] + j * 16];
      o[j] = wmma_f16(ap, bv, o[j]);
    }
    __syncthreads();
  }

  // epilogue: O / l, store valid query rows (contiguous node order)
#pragma unroll
  for (int j = 0; j < 4; ++j)
#pragma unroll
    for (int r = 0; r < 8; ++r) {
      int m  = r + (hih ? 8 : 0);
      int qq = qt * 16 + m;
      if (qq < L)
        oatt[(size_t)(s + qq) * EE + h * DHD + j * 16 + nl] =
            (_Float16)(o[j][r] / lrow[r]);
    }
}

// ---------------------------------------------------------------------------
// 6) out_proj (WMMA, 32 rows x full 768 cols per block) fused with SiLU MLP.
//    Block = 8 waves; wave w owns cols w*96..w*96+95 with 2 row-tiles.
//    Staged activations kept in f16 LDS (48KB).  grid.x = N/32
// ---------------------------------------------------------------------------
__global__ __launch_bounds__(256) void k_out_mlp(
    const _Float16* __restrict__ Ao, const _Float16* __restrict__ Wo,
    const float* __restrict__ bo, const float* __restrict__ w1,
    const float* __restrict__ b1, const float* __restrict__ w2,
    const float* __restrict__ b2, float* __restrict__ out) {
  __shared__ __align__(32) _Float16 XO[32][EE];  // 48KB
  __shared__ float Hh[32][HIDN];                 // 2KB
  int tid = threadIdx.x;
  int lane = tid & 31, wv = tid >> 5;   // 8 waves
  int row0 = blockIdx.x * 32;
  int nl = lane & 15;
  bool hih = lane >= 16;
  int k0 = hih ? 16 : 0, kb = hih ? 16 : 0;
  int jb = wv * 96;

  v8f acc[2][6];
#pragma unroll
  for (int mt = 0; mt < 2; ++mt)
#pragma unroll
    for (int t = 0; t < 6; ++t) acc[mt][t] = zero8f();

  for (int kc = 0; kc < EE; kc += 32) {
    v16h a[2];
#pragma unroll
    for (int mt = 0; mt < 2; ++mt)
      a[mt] = load_afrag(Ao + (size_t)(row0 + mt * 16 + nl) * EE + kc + k0);
#pragma unroll
    for (int t = 0; t < 6; ++t) {
      int j = jb + t * 16 + nl;
      v16h b = load_bfrag(Wo + (size_t)j * EE + kc + kb);
#pragma unroll
      for (int mt = 0; mt < 2; ++mt)
        acc[mt][t] = wmma_f16(a[mt], b, acc[mt][t]);
    }
  }
#pragma unroll
  for (int t = 0; t < 6; ++t) {
    int j = jb + t * 16 + nl;
    float bj = bo[j];
#pragma unroll
    for (int mt = 0; mt < 2; ++mt)
#pragma unroll
      for (int r = 0; r < 8; ++r)
        XO[mt * 16 + r + (hih ? 8 : 0)][j] = (_Float16)(acc[mt][t][r] + bj);
  }
  __syncthreads();

  // MLP layer 1 + SiLU: 32 rows x 16 hidden = 512 dots, 2 per thread
#pragma unroll
  for (int pp = 0; pp < 2; ++pp) {
    int idx = tid + pp * 256;
    int rr = idx >> 4, ii = idx & 15;
    const float* w1r = w1 + ii * EE;
    float sum = b1[ii];
    for (int e = 0; e < EE; ++e) sum += (float)XO[rr][e] * w1r[e];
    Hh[rr][ii] = sum / (1.0f + __expf(-sum));   // SiLU
  }
  __syncthreads();

  // MLP layer 2: out[row, e] = sum_j Hh[row][j] * w2[e][j] + b2[e]
  for (int idx = tid; idx < 32 * EE; idx += 256) {
    int rr = idx / EE, e = idx - rr * EE;
    float sum = b2[e];
#pragma unroll
    for (int j = 0; j < HIDN; ++j) sum += Hh[rr][j] * w2[e * HIDN + j];
    out[(size_t)(row0 + rr) * EE + e] = sum;
  }
}

// ---------------------------------------------------------------------------
extern "C" void kernel_launch(void* const* d_in, const int* in_sizes, int n_in,
                              void* d_out, int out_size, void* d_ws,
                              size_t ws_size, hipStream_t stream) {
  const float* x          = (const float*)d_in[0];
  const float* ln_w       = (const float*)d_in[1];
  const float* ln_b       = (const float*)d_in[2];
  const float* in_proj_w  = (const float*)d_in[3];
  const float* in_proj_b  = (const float*)d_in[4];
  const float* out_proj_w = (const float*)d_in[5];
  const float* out_proj_b = (const float*)d_in[6];
  const float* w1         = (const float*)d_in[7];
  const float* b1         = (const float*)d_in[8];
  const float* w2         = (const float*)d_in[9];
  const float* b2         = (const float*)d_in[10];
  const int*   batch      = (const int*)d_in[11];

  char* p = (char*)d_ws;
  int* counts = (int*)p;              p += 256;
  int* starts = (int*)p;              p += 256;
  _Float16* xn   = (_Float16*)p;      p += (size_t)NN * EE * 2;
  _Float16* qkv  = (_Float16*)p;      p += (size_t)NN * J3E * 2;
  _Float16* oatt = (_Float16*)p;      p += (size_t)NN * EE * 2;
  _Float16* wqh  = (_Float16*)p;      p += (size_t)J3E * EE * 2;
  _Float16* woh  = (_Float16*)p;      p += (size_t)EE * EE * 2;

  k_starts<<<1, 64, 0, stream>>>(batch, NN, starts, counts);
  k_ln<<<NN, 256, 0, stream>>>(x, ln_w, ln_b, xn);
  k_f2h<<<1024, 256, 0, stream>>>(in_proj_w, wqh, J3E * EE);
  k_f2h<<<512, 256, 0, stream>>>(out_proj_w, woh, EE * EE);
  k_gemm_bias<<<dim3(NN / 64, J3E / 256), 128, 0, stream>>>(
      xn, wqh, in_proj_b, qkv, EE, J3E);
  k_attn<<<dim3(32, HH, BBG), 32, 0, stream>>>(qkv, starts, counts, oatt);
  k_out_mlp<<<NN / 32, 256, 0, stream>>>(oatt, woh, out_proj_b, w1, b1, w2, b2,
                                         (float*)d_out);
}
